// EfficientMultiHeadLocalSelfAttention_70531952934981
// MI455X (gfx1250) — matrix operand order repaired and checked by reading, not compiled, not run on hardware
//
#include <hip/hip_runtime.h>
#include <hip/hip_bf16.h>
#include <stdint.h>

// ---------------------------------------------------------------------------
// Problem constants (from the reference): B=2, T=2048, D=512, H=8, d=64, W=32
// ---------------------------------------------------------------------------
#define TT   2048
#define BB   2
#define DD   512
#define HH   8
#define HD   64          // head dim
#define WW   32          // half window
#define NK   80          // key span per 16-query block: 16 + 2*32
#define NKP  96          // NK padded to multiple of 32 (WMMA K step)
#define MT   (BB*TT)     // 4096 rows in the GEMM view

typedef __attribute__((ext_vector_type(16))) __bf16        v16bf;
typedef __attribute__((ext_vector_type(8)))  float         v8f;
typedef __attribute__((ext_vector_type(4)))  unsigned int  v4u;
typedef __attribute__((ext_vector_type(8)))  unsigned int  v8u;

#define DI __device__ __forceinline__

DI unsigned short f2bf(float f) {
  unsigned int u = __float_as_uint(f);
  unsigned int r = (u + 0x7FFFu + ((u >> 16) & 1u)) >> 16;   // round-nearest-even
  return (unsigned short)r;
}

// Fragment overlay: 16 bf16 = 32 bytes = 8 VGPRs per lane.
union Frag16 {
  v16bf          v;
  unsigned short h[16];
  uint4          q[2];
};

// ---------------------------------------------------------------------------
// Tensor Data Mover: issue a 2D bf16 tile load (tile_d0 x tile_d1, row-major)
// from global into LDS. D# built per CDNA5 ISA (group0 128b, group1 256b),
// issued with inline asm (portable across ROCm7.2 / amdgpu-toolchain).
// Tracked by TENSORcnt. EXEC is ignored; call from ONE wave per workgroup.
// ---------------------------------------------------------------------------
DI void tdm_load_tile_2d(unsigned lds_addr, const unsigned short* gaddr,
                         unsigned tensor_d0, unsigned tensor_d1,
                         unsigned tile_d0, unsigned tile_d1,
                         unsigned stride0_elems) {
  unsigned long long ga = (unsigned long long)gaddr;
  v4u g0;
  g0.x = 1u;                                              // count=1, user D#
  g0.y = lds_addr;                                        // LDS byte address
  g0.z = (unsigned)ga;                                    // global_addr[31:0]
  g0.w = (unsigned)((ga >> 32) & 0x01FFFFFFu) | (2u << 30); // addr[56:32], type=2
  v8u g1;
  g1.s0 = 1u << 16;                                       // data_size=1 (2 bytes)
  g1.s1 = (tensor_d0 & 0xFFFFu) << 16;                    // tensor_dim0[15:0] @63:48
  g1.s2 = (tensor_d0 >> 16) | ((tensor_d1 & 0xFFFFu) << 16); // dim0 hi | dim1 lo
  g1.s3 = (tensor_d1 >> 16) | (tile_d0 << 16);            // dim1 hi | tile_dim0
  g1.s4 = tile_d1 & 0xFFFFu;                              // tile_dim1 (tile_dim2=0)
  g1.s5 = stride0_elems;                                  // tensor_dim0_stride lo
  g1.s6 = 0u;                                             // stride hi / dim1_stride lo
  g1.s7 = 0u;
  asm volatile("tensor_load_to_lds %0, %1" :: "s"(g0), "s"(g1) : "memory");
}

// ---------------------------------------------------------------------------
// f32 -> bf16 conversion pass (for x)
// ---------------------------------------------------------------------------
__global__ void cvt_f32_to_bf16(const float* __restrict__ in,
                                unsigned short* __restrict__ out, int n) {
  int i = blockIdx.x * blockDim.x + threadIdx.x;
  if (i < n) out[i] = f2bf(in[i]);
}

// ---------------------------------------------------------------------------
// f32 (KxN) -> bf16 transposed (NxK), 32x32 LDS tiles (256 threads)
// Transposed weights make every GEMM B tile a row-major 64x32 block, so both
// TDM tile loads and ds_load_b128 fragment reads apply to A and B alike.
// ---------------------------------------------------------------------------
__global__ __launch_bounds__(256)
void cvt_transpose_bf16(const float* __restrict__ in,
                        unsigned short* __restrict__ out, int K, int N) {
  __shared__ float tile[32][33];
  const int k0 = blockIdx.x * 32, n0 = blockIdx.y * 32;
  const int tx = threadIdx.x & 31, ty = threadIdx.x >> 5;
  for (int i = ty; i < 32; i += 8)
    tile[i][tx] = in[(size_t)(k0 + i) * N + n0 + tx];
  __syncthreads();
  for (int i = ty; i < 32; i += 8)
    out[(size_t)(n0 + i) * K + k0 + tx] = f2bf(tile[tx][i]);
}

// ---------------------------------------------------------------------------
// bf16 WMMA GEMM with TDM-fed LDS tiles:
//   C(MxN) = A(MxK) @ Bt(NxK)^T + bias
// 128 threads (4 waves). Macro tile 64x64; each wave owns a 32x32 quadrant
// (2x2 v_wmma_f32_16x16x32_bf16). K-panel 32. Wave 0 drives the TDM.
// ---------------------------------------------------------------------------
template <bool BF16OUT>
__global__ __launch_bounds__(128)
void gemm_bf16_wmma_tdm(const unsigned short* __restrict__ A,
                        const unsigned short* __restrict__ Bt,
                        const float* __restrict__ bias,
                        void* __restrict__ outp, int M, int N, int K) {
  __shared__ unsigned short As[64][32];   // A tile  [m][k], row-major
  __shared__ unsigned short Bs[64][32];   // B tile  [n][k], row-major (pre-transposed)

  const int tid   = threadIdx.x;
  const int lane  = tid & 31;
  const int wave  = tid >> 5;
  const int l16   = lane & 15;
  const int khalf = lane >> 4;            // lane[4]: K-half select (ISA 7.12.2)
  const int m0 = blockIdx.y * 64;
  const int n0 = blockIdx.x * 64;
  const int wm = (wave >> 1) * 32;        // wave quadrant inside macro tile
  const int wn = (wave & 1) * 32;

  const unsigned ldsA = (unsigned)(unsigned long long)(void*)&As[0][0];
  const unsigned ldsB = (unsigned)(unsigned long long)(void*)&Bs[0][0];

  v8f acc[2][2];
  for (int i = 0; i < 2; ++i)
    for (int j = 0; j < 2; ++j) acc[i][j] = (v8f)0.0f;

  for (int kt = 0; kt < K; kt += 32) {
    // --- TDM: async tile loads into LDS (wave 0 only; EXEC-independent) ---
    if (wave == 0) {
      tdm_load_tile_2d(ldsA, A  + (size_t)m0 * K + kt, K, M, 32, 64, K);
      tdm_load_tile_2d(ldsB, Bt + (size_t)n0 * K + kt, K, N, 32, 64, K);
      __builtin_amdgcn_s_wait_tensorcnt(0);
    }
    __syncthreads();                      // LDS tiles valid for all 4 waves

    // --- A fragments: lane row m=l16; two contiguous 8-K runs -> 2x b128 ---
    Frag16 af[2];
    for (int i = 0; i < 2; ++i) {
      int m = wm + 16 * i + l16;
      af[i].q[0] = *(const uint4*)&As[m][8 * khalf];
      af[i].q[1] = *(const uint4*)&As[m][16 + 8 * khalf];
    }
    // --- B fragments: lane col n=l16; two contiguous 8-K runs -> 2x b128 ---
    Frag16 bfr[2];
    for (int j = 0; j < 2; ++j) {
      int n = wn + 16 * j + l16;
      bfr[j].q[0] = *(const uint4*)&Bs[n][8 * khalf];
      bfr[j].q[1] = *(const uint4*)&Bs[n][16 + 8 * khalf];
    }
    for (int i = 0; i < 2; ++i)
      for (int j = 0; j < 2; ++j)
        acc[i][j] = __builtin_amdgcn_wmma_f32_16x16x32_bf16(
            false, af[i].v, false, bfr[j].v, (short)0, acc[i][j], false, false);
    __syncthreads();                      // protect LDS from next TDM write
  }

  // --- epilogue (branch resolved at compile time): D(m,n): m=v+8*khalf, n=l16
  for (int i = 0; i < 2; ++i)
    for (int j = 0; j < 2; ++j)
      for (int v = 0; v < 8; ++v) {
        int m = m0 + wm + 16 * i + v + 8 * khalf;
        int n = n0 + wn + 16 * j + l16;
        float val = acc[i][j][v] + bias[n];
        if constexpr (BF16OUT)
          ((unsigned short*)outp)[(size_t)m * N + n] = f2bf(val);
        else
          ((float*)outp)[(size_t)m * N + n] = val;
      }
}

// ---------------------------------------------------------------------------
// Local windowed attention, one wave (32 threads) per (b, h, 16-query block).
// Scores 16x80 via 10 WMMAs, f32 softmax in LDS, AV 16x64 via 12 WMMAs.
// ---------------------------------------------------------------------------
__global__ __launch_bounds__(32)
void local_attn_wmma(const unsigned short* __restrict__ Qb,
                     const unsigned short* __restrict__ Kb,
                     const unsigned short* __restrict__ Vb,
                     unsigned short* __restrict__ Ab) {
  __shared__ float          S[16][NK];    // raw scores (f32)
  __shared__ unsigned short P[16][NKP];   // softmax probs (bf16), zero-padded K
  __shared__ unsigned short Vt[HD][NKP];  // V window transposed: [dim][key]

  const int lane  = threadIdx.x;
  const int l16   = lane & 15;
  const int khalf = lane >> 4;
  const int tb = blockIdx.x;              // query block
  const int h  = blockIdx.y;
  const int bz = blockIdx.z;
  const int t0 = tb * 16;

  // ---- Q fragments: rows m = l16 -> query t0+m, 2 K-steps over head dim ----
  const size_t qrow = ((size_t)(bz * TT + t0 + l16)) * DD + h * HD;
  Frag16 qa[2];
  for (int s = 0; s < 2; ++s) {
    qa[s].q[0] = *(const uint4*)(Qb + qrow + 32 * s + 8 * khalf);
    qa[s].q[1] = *(const uint4*)(Qb + qrow + 32 * s + 16 + 8 * khalf);
  }

  // ---- scores: 5 key tiles of 16, keys t0-32 .. t0+47 ----
  v8f sacc[5];
  for (int g = 0; g < 5; ++g) sacc[g] = (v8f)0.0f;
  for (int g = 0; g < 5; ++g) {
    int key = t0 - WW + 16 * g + l16;     // B-matrix column n = l16 -> key
    bool valid = (key >= 0) && (key < TT);
    for (int s = 0; s < 2; ++s) {
      Frag16 kb;
      if (valid) {
        const unsigned short* src =
            Kb + ((size_t)(bz * TT + key)) * DD + h * HD + 32 * s + 8 * khalf;
        kb.q[0] = *(const uint4*)src;          // two 8-dim runs (K rows)
        kb.q[1] = *(const uint4*)(src + 16);
      } else {
        kb.q[0] = make_uint4(0u, 0u, 0u, 0u);
        kb.q[1] = make_uint4(0u, 0u, 0u, 0u);
      }
      sacc[g] = __builtin_amdgcn_wmma_f32_16x16x32_bf16(
          false, qa[s].v, false, kb.v, (short)0, sacc[g], false, false);
    }
  }
  // spill scores: element (m, n): m = v + 8*khalf, n = 16*g + l16
  for (int g = 0; g < 5; ++g)
    for (int v = 0; v < 8; ++v)
      S[v + 8 * khalf][16 * g + l16] = sacc[g][v];
  __syncthreads();

  // ---- softmax (lanes 0..15, one row each) ----
  if (lane < 16) {
    const int m = lane, t = t0 + m;
    float mx = -1e30f;
    for (int n = 0; n < NK; ++n) {
      int key = t0 - WW + n;
      int rel = key - t;
      bool valid = (key >= 0) && (key < TT) && (rel >= -WW) && (rel <= WW);
      float sc = valid ? S[m][n] * 0.125f : -1e30f;   // 1/sqrt(64)
      S[m][n] = sc;
      mx = fmaxf(mx, sc);
    }
    float sum = 0.f;
    for (int n = 0; n < NK; ++n) {
      float e = (S[m][n] > -1e29f) ? __expf(S[m][n] - mx) : 0.f;
      S[m][n] = e;
      sum += e;
    }
    float inv = 1.f / sum;                             // diagonal always valid
    for (int n = 0; n < NK; ++n)  P[m][n] = f2bf(S[m][n] * inv);
    for (int n = NK; n < NKP; ++n) P[m][n] = 0;
  }
  // ---- stage V window transposed (all lanes, vector global reads) ----
  for (int r = lane; r < NKP; r += 32) {
    int key = t0 - WW + r;
    unsigned short buf[HD];
    if (r < NK && key >= 0 && key < TT) {
      const uint4* src = (const uint4*)(Vb + ((size_t)(bz * TT + key)) * DD + h * HD);
#pragma unroll
      for (int i = 0; i < HD / 8; ++i) ((uint4*)buf)[i] = src[i];
    } else {
#pragma unroll
      for (int i = 0; i < HD; ++i) buf[i] = 0;
    }
#pragma unroll
    for (int dim = 0; dim < HD; ++dim) Vt[dim][r] = buf[dim];
  }
  __syncthreads();

  // ---- O = P(16x96) @ Vwin(96x64): 3 K-steps x 4 N-tiles ----
  v8f oacc[4];
  for (int nt = 0; nt < 4; ++nt) oacc[nt] = (v8f)0.0f;
  for (int s = 0; s < 3; ++s) {
    Frag16 pa;                                         // A frag from P (LDS)
    pa.q[0] = *(const uint4*)&P[l16][32 * s + 8 * khalf];
    pa.q[1] = *(const uint4*)&P[l16][32 * s + 16 + 8 * khalf];
    for (int nt = 0; nt < 4; ++nt) {
      Frag16 vb;                                       // B frag from Vt (LDS)
      vb.q[0] = *(const uint4*)&Vt[16 * nt + l16][32 * s + 8 * khalf];
      vb.q[1] = *(const uint4*)&Vt[16 * nt + l16][32 * s + 16 + 8 * khalf];
      oacc[nt] = __builtin_amdgcn_wmma_f32_16x16x32_bf16(
          false, pa.v, false, vb.v, (short)0, oacc[nt], false, false);
    }
  }
  for (int nt = 0; nt < 4; ++nt)
    for (int v = 0; v < 8; ++v) {
      int m = v + 8 * khalf;
      Ab[((size_t)(bz * TT + t0 + m)) * DD + h * HD + 16 * nt + l16] =
          f2bf(oacc[nt][v]);
    }
}

// ---------------------------------------------------------------------------
// Host-side orchestration
// ---------------------------------------------------------------------------
extern "C" void kernel_launch(void* const* d_in, const int* in_sizes, int n_in,
                              void* d_out, int out_size, void* d_ws, size_t ws_size,
                              hipStream_t stream) {
  const float* x  = (const float*)d_in[0];
  const float* wq = (const float*)d_in[1];
  const float* bq = (const float*)d_in[2];
  const float* wk = (const float*)d_in[3];
  const float* bk = (const float*)d_in[4];
  const float* wv = (const float*)d_in[5];
  const float* bv = (const float*)d_in[6];
  const float* wo = (const float*)d_in[7];
  const float* bo = (const float*)d_in[8];
  float* out = (float*)d_out;

  // Workspace layout (bf16/ushort units). Weights stored TRANSPOSED (NxK).
  // Ab reuses xb (x retired after QKV). Total ws use ~18 MB.
  const size_t NX = (size_t)MT * DD;     // 4096*512
  const size_t NW = (size_t)DD * DD;     // 512*512
  unsigned short* xb  = (unsigned short*)d_ws;
  unsigned short* wqb = xb  + NX;
  unsigned short* wkb = wqb + NW;
  unsigned short* wvb = wkb + NW;
  unsigned short* wob = wvb + NW;
  unsigned short* Qb  = wob + NW;
  unsigned short* Kb  = Qb  + NX;
  unsigned short* Vb  = Kb  + NX;
  unsigned short* Ab  = xb;              // alias

  // 1) down-convert x (linear) and weights (transposed) to bf16
  cvt_f32_to_bf16<<<(int)((NX + 255) / 256), 256, 0, stream>>>(x, xb, (int)NX);
  dim3 gt(DD / 32, DD / 32);
  cvt_transpose_bf16<<<gt, 256, 0, stream>>>(wq, wqb, DD, DD);
  cvt_transpose_bf16<<<gt, 256, 0, stream>>>(wk, wkb, DD, DD);
  cvt_transpose_bf16<<<gt, 256, 0, stream>>>(wv, wvb, DD, DD);
  cvt_transpose_bf16<<<gt, 256, 0, stream>>>(wo, wob, DD, DD);

  // 2) Q/K/V projections (bf16 out), TDM-fed WMMA GEMM
  dim3 gg(DD / 64, MT / 64);
  gemm_bf16_wmma_tdm<true><<<gg, 128, 0, stream>>>(xb, wqb, bq, Qb, MT, DD, DD);
  gemm_bf16_wmma_tdm<true><<<gg, 128, 0, stream>>>(xb, wkb, bk, Kb, MT, DD, DD);
  gemm_bf16_wmma_tdm<true><<<gg, 128, 0, stream>>>(xb, wvb, bv, Vb, MT, DD, DD);

  // 3) local attention (one wave per (b,h,16-query block))
  dim3 ga(TT / 16, HH, BB);
  local_attn_wmma<<<ga, 32, 0, stream>>>(Qb, Kb, Vb, Ab);

  // 4) output projection (f32 out, + bias)
  gemm_bf16_wmma_tdm<false><<<gg, 128, 0, stream>>>(Ab, wob, bo, out, MT, DD, DD);
}
